// Qwen3_5Attention_21440476742417
// MI455X (gfx1250) — matrix operand order repaired and checked by reading, not compiled, not run on hardware
//
#include <hip/hip_runtime.h>
#include <hip/hip_bf16.h>
#include <math.h>

typedef __bf16 bf16_t;
typedef __bf16 v16bf __attribute__((ext_vector_type(16)));
typedef float  v8f   __attribute__((ext_vector_type(8)));
typedef unsigned int u32x4 __attribute__((ext_vector_type(4)));
typedef int   i32x4 __attribute__((ext_vector_type(4)));
typedef int   i32x8 __attribute__((ext_vector_type(8)));

#define B_ 2
#define S_ 2048
#define M_ 2048
#define H_ 16
#define KV_ 4
#define HD_ 128
#define GROUPS_ (H_ / KV_)
#define EPS_ 1e-6f
#define NEG_INF_ (-1.0e9f)

union Frag {
  v16bf v;
  uint4 u[2];
};

__device__ inline v8f vzero8() { v8f v = {}; return v; }

__device__ inline float red_max16(float v) {
  v = fmaxf(v, __shfl_xor(v, 8, 32));
  v = fmaxf(v, __shfl_xor(v, 4, 32));
  v = fmaxf(v, __shfl_xor(v, 2, 32));
  v = fmaxf(v, __shfl_xor(v, 1, 32));
  return v;
}
__device__ inline float red_sum16(float v) {
  v += __shfl_xor(v, 8, 32);
  v += __shfl_xor(v, 4, 32);
  v += __shfl_xor(v, 2, 32);
  v += __shfl_xor(v, 1, 32);
  return v;
}

// ---------------------------------------------------------------- TDM (async tensor)
// 2D tensor_load_to_lds: builds D# group0/group1 per CDNA5 ISA 8.3/8.4.
// data_size = 2 bytes (bf16). Optional LDS pad: +4 DWORDs after every 64 DWORDs
// (i.e. +8 bf16 after every 128 -> padded row stride 136 elems).
#if __has_include(<hip/amd_detail/amd_gfx1250_TDM.h>)
#define TDM_6ARG 1
#endif

__device__ inline void tdm_load_2d(unsigned lds_addr, const void* gptr,
                                   unsigned tensor_d0, unsigned tensor_d1,
                                   unsigned tile_d0, unsigned tile_d1,
                                   unsigned long long stride0_elems, bool pad) {
  unsigned long long ga = (unsigned long long)(size_t)gptr;
  u32x4 g0;
  g0[0] = 1u;  // count=1, user descriptor, no gather
  g0[1] = lds_addr;
  g0[2] = (unsigned)(ga & 0xFFFFFFFFu);
  g0[3] = (unsigned)((ga >> 32) & 0x01FFFFFFu) | (2u << 30);  // type=2 ("image")
  i32x8 g1;
  unsigned w0 = (1u << 16);  // data_size = 1 -> 2 bytes
  if (pad) w0 |= (1u << 20) | (5u << 22) | (3u << 25);  // pad_en, interval=64dw, amount=4dw
  g1[0] = (int)w0;
  g1[1] = (int)((tensor_d0 & 0xFFFFu) << 16);                                  // dim0[15:0]
  g1[2] = (int)(((tensor_d0 >> 16) & 0xFFFFu) | ((tensor_d1 & 0xFFFFu) << 16));// dim0[31:16]|dim1[15:0]
  g1[3] = (int)(((tensor_d1 >> 16) & 0xFFFFu) | ((tile_d0 & 0xFFFFu) << 16));  // dim1[31:16]|tile0
  g1[4] = (int)(tile_d1 & 0xFFFFu);                                            // tile1 (tile2=0)
  g1[5] = (int)(unsigned)(stride0_elems & 0xFFFFFFFFull);
  g1[6] = (int)(unsigned)((stride0_elems >> 32) & 0xFFFFull);
  g1[7] = 0;
  i32x4 z4 = {};
#ifdef TDM_6ARG
  i32x8 z8 = {};
  __builtin_amdgcn_tensor_load_to_lds(g0, g1, z4, z4, z8, 0);
#else
  __builtin_amdgcn_tensor_load_to_lds(g0, g1, z4, z4, 0);
#endif
}

// ---------------------------------------------------------------- fp32 -> bf16
__global__ __launch_bounds__(256) void cvt_f32_bf16(const float* __restrict__ src,
                                                    bf16_t* __restrict__ dst, int n) {
  int i = blockIdx.x * blockDim.x + threadIdx.x;
  int stride = gridDim.x * blockDim.x;
  for (; i < n; i += stride) dst[i] = (bf16_t)src[i];
}

// ---------------------------------------------------------------- bf16 WMMA GEMM
// C[Mrows x N] = A[Mrows x K] * W[K x N]; block tile 128x256, 8 waves (2x4),
// each wave computes 64x64 = 4x4 fragments of 16x16; K stepped by 32.
// A tile staged by the Tensor Data Mover; W tile staged transposed by the block.
template <bool OUT_BF16>
__global__ __launch_bounds__(256) void gemm_bf16_wmma(const bf16_t* __restrict__ A,
                                                      const bf16_t* __restrict__ W,
                                                      void* __restrict__ Cvoid,
                                                      int Mrows, int N, int K) {
  __shared__ __align__(16) bf16_t As[128][32];   // [m][k]  (TDM destination)
  __shared__ __align__(16) bf16_t Bs[256][32];   // [n][k]  (transposed stage)

  const int tid = threadIdx.x;
  const int lane = tid & 31;
  const int wave = tid >> 5;
  const int waveM = wave >> 2;   // 0..1
  const int waveN = wave & 3;    // 0..3
  const int m0 = blockIdx.y * 128;
  const int n0 = blockIdx.x * 256;

  v8f acc[4][4];
#pragma unroll
  for (int i = 0; i < 4; i++)
#pragma unroll
    for (int j = 0; j < 4; j++) acc[i][j] = vzero8();

  const int lb_krow = tid >> 3;          // 0..31
  const int lb_ncol = (tid & 7) * 32;    // 0..224

  const int rsel = lane & 15;
  const int kloA = (lane & 16) ? 16 : 0;
  const int kloB = (lane & 16) ? 16 : 0;
  const unsigned asAddr = (unsigned)(size_t)&As[0][0];

  for (int k0 = 0; k0 < K; k0 += 32) {
    // async A tile: 128 rows x 32 cols, row stride K (TDM, wave 0 issues)
    if (wave == 0)
      tdm_load_2d(asAddr, A + (size_t)m0 * K + k0,
                  (unsigned)(K - k0), (unsigned)(Mrows - m0), 32u, 128u,
                  (unsigned long long)K, false);
    // stage W tile transposed -> Bs[n][k] (32 rows x 256 cols)
    {
      const bf16_t* wg = W + (size_t)(k0 + lb_krow) * N + n0 + lb_ncol;
#pragma unroll
      for (int u = 0; u < 4; u++) {
        Frag wf;
        wf.u[0] = *(const uint4*)(wg + u * 8);
#pragma unroll
        for (int i = 0; i < 8; i++) Bs[lb_ncol + u * 8 + i][lb_krow] = wf.v[i];
      }
      if (k0 + 32 < K) __builtin_prefetch(wg + (size_t)32 * N, 0, 1);  // global_prefetch
    }
    __builtin_amdgcn_s_wait_tensorcnt((short)0);
    __syncthreads();

    // A fragments: lane<16 holds k 0..15, lane>=16 holds k 16..31 (contiguous)
    Frag af[4];
#pragma unroll
    for (int mt = 0; mt < 4; mt++) {
      const bf16_t* p = &As[waveM * 64 + mt * 16 + rsel][0];
      af[mt].u[0] = *(const uint4*)(p + kloA);
      af[mt].u[1] = *(const uint4*)(p + kloA + 8);
    }
    // B fragments: lane<16 holds k 0..15, lane>=16 holds k 16..31 (contiguous)
    Frag bfr[4];
#pragma unroll
    for (int nt = 0; nt < 4; nt++) {
      const bf16_t* p = &Bs[waveN * 64 + nt * 16 + rsel][0];
      bfr[nt].u[0] = *(const uint4*)(p + kloB);
      bfr[nt].u[1] = *(const uint4*)(p + kloB + 8);
    }
#pragma unroll
    for (int mt = 0; mt < 4; mt++)
#pragma unroll
      for (int nt = 0; nt < 4; nt++)
        acc[mt][nt] = __builtin_amdgcn_wmma_f32_16x16x32_bf16(
            false, af[mt].v, false, bfr[nt].v, (short)0, acc[mt][nt], false, false);
    __syncthreads();
  }

  const int rowOff = (lane >> 4) << 3;
#pragma unroll
  for (int mt = 0; mt < 4; mt++)
#pragma unroll
    for (int nt = 0; nt < 4; nt++)
#pragma unroll
      for (int r = 0; r < 8; r++) {
        const int gm = m0 + waveM * 64 + mt * 16 + rowOff + r;
        const int gn = n0 + waveN * 64 + nt * 16 + rsel;
        const float v = acc[mt][nt][r];
        if (OUT_BF16)
          ((bf16_t*)Cvoid)[(size_t)gm * N + gn] = (bf16_t)v;
        else
          ((float*)Cvoid)[(size_t)gm * N + gn] = v;
      }
}

// ------------------------------------------------- RMS-norm + partial RoPE (+gate)
template <bool GATE>
__global__ __launch_bounds__(256) void epi_norm_rope(bf16_t* __restrict__ data,
                                                     const float* __restrict__ scale,
                                                     int nrows, int rowStride, int heads) {
  const int lane = threadIdx.x & 31;
  const int wave = threadIdx.x >> 5;
  const int row = blockIdx.x * 8 + wave;
  if (row >= nrows) return;
  const int spos = (row / heads) % S_;
  bf16_t* p = data + (size_t)row * rowStride;

  float q[4];
#pragma unroll
  for (int j = 0; j < 4; j++) q[j] = (float)p[lane + 32 * j];
  float ss = q[0] * q[0] + q[1] * q[1] + q[2] * q[2] + q[3] * q[3];
#pragma unroll
  for (int m = 16; m >= 1; m >>= 1) ss += __shfl_xor(ss, m, 32);
  const float rinv = rsqrtf(ss * (1.0f / 128.0f) + EPS_);
#pragma unroll
  for (int j = 0; j < 4; j++) q[j] = q[j] * rinv * scale[lane + 32 * j];
  // partial RoPE over d = 0..31 (j == 0): pair (d, d^16) == lane xor 16
  {
    const float x = q[0];
    const float partner = __shfl_xor(x, 16, 32);
    const int i = lane & 15;
    const float f = (float)spos * __powf(10000.0f, -(float)i * (1.0f / 16.0f));
    const float c = __cosf(f), s = __sinf(f);
    q[0] = (lane < 16) ? (x * c - partner * s) : (x * c + partner * s);
  }
#pragma unroll
  for (int j = 0; j < 4; j++) p[lane + 32 * j] = (bf16_t)q[j];

  if (GATE) {
#pragma unroll
    for (int j = 0; j < 4; j++) {
      const float g = (float)p[128 + lane + 32 * j];
      p[128 + lane + 32 * j] = (bf16_t)(1.0f / (1.0f + __expf(-g)));
    }
  }
}

// -------------------------------------------------------- flash attention (bf16 WMMA)
// grid (S/64, H, B); 4 waves/block, wave = one 16-row q tile; waves share K/V
// LDS tiles of 32 kv rows; K tile staged by TDM (with LDS padding to 136/row).
__global__ __launch_bounds__(128) void attn_fwd(const bf16_t* __restrict__ qg,  // [B,S,H,256]
                                                const bf16_t* __restrict__ kk,  // [B,S,KV,128]
                                                const bf16_t* __restrict__ vv,  // [B,S,KV,128]
                                                bf16_t* __restrict__ out) {     // [B,S,H*128]
  __shared__ __align__(16) bf16_t Ks[32][136];       // [kv][d], TDM-padded rows
  __shared__ __align__(16) bf16_t Vt[128][40];       // [d][kv] transposed
  __shared__ __align__(16) bf16_t Pls[4][16][32];    // per-wave P bounce

  const int lane = threadIdx.x & 31;
  const int wave = threadIdx.x >> 5;
  const int h = blockIdx.y;
  const int b = blockIdx.z;
  const int kvh = h / GROUPS_;
  const int q0 = blockIdx.x * 64 + wave * 16;
  const int rsel = lane & 15;
  const int rowOff = (lane >> 4) << 3;
  const float inv_norm = 0.08838834764831845f;  // 1/sqrt(128)
  const unsigned ksAddr = (unsigned)(size_t)&Ks[0][0];

  // Q fragments (4 chunks of K=32 over HD)
  Frag qf[4];
  {
    const bf16_t* qp = qg + (((size_t)(b * S_ + q0 + rsel) * H_) + h) * 256;
    const int klo = (lane & 16) ? 16 : 0;
#pragma unroll
    for (int c = 0; c < 4; c++) {
      qf[c].u[0] = *(const uint4*)(qp + c * 32 + klo);
      qf[c].u[1] = *(const uint4*)(qp + c * 32 + klo + 8);
    }
  }

  v8f o[8];
#pragma unroll
  for (int f = 0; f < 8; f++) o[f] = vzero8();
  float mrow[8], lrow[8];
#pragma unroll
  for (int r = 0; r < 8; r++) { mrow[r] = -3.0e38f; lrow[r] = 0.0f; }

  const int t = threadIdx.x;
  const int lrowK = t >> 2;          // 0..31
  const int lcolK = (t & 3) * 32;    // 0,32,64,96
  const int nkt = blockIdx.x * 2 + 2;

  for (int jt = 0; jt < nkt; ++jt) {
    const int kbase = jt * 32;
    // async K tile via TDM: 32 rows x 128 cols, row stride KV*128, pad to 136
    if (wave == 0)
      tdm_load_2d(ksAddr,
                  kk + (((size_t)(b * S_ + kbase) * KV_) + kvh) * 128,
                  128u, 32u, 128u, 32u,
                  (unsigned long long)(KV_ * 128), true);
    // stage V transposed (manual: TDM cannot transpose)
    {
      const bf16_t* vg = vv + (((size_t)(b * S_ + kbase + lrowK) * KV_) + kvh) * 128 + lcolK;
#pragma unroll
      for (int u = 0; u < 4; u++) {
        Frag vf;
        vf.u[0] = *(const uint4*)(vg + u * 8);
#pragma unroll
        for (int i = 0; i < 8; i++) Vt[lcolK + u * 8 + i][lrowK] = vf.v[i];
      }
    }
    __builtin_amdgcn_s_wait_tensorcnt((short)0);
    __syncthreads();

    // scores: two 16x16 kv halves, contraction over HD in 4 chunks of 32
    v8f s0 = vzero8(), s1 = vzero8();
    {
      const int dlo = (lane & 16) ? 16 : 0;
#pragma unroll
      for (int c = 0; c < 4; c++) {
        Frag bk0, bk1;
        const bf16_t* p0 = &Ks[rsel][0] + c * 32 + dlo;
        const bf16_t* p1 = &Ks[rsel + 16][0] + c * 32 + dlo;
        bk0.u[0] = *(const uint4*)(p0); bk0.u[1] = *(const uint4*)(p0 + 8);
        bk1.u[0] = *(const uint4*)(p1); bk1.u[1] = *(const uint4*)(p1 + 8);
        s0 = __builtin_amdgcn_wmma_f32_16x16x32_bf16(false, qf[c].v, false, bk0.v, (short)0, s0, false, false);
        s1 = __builtin_amdgcn_wmma_f32_16x16x32_bf16(false, qf[c].v, false, bk1.v, (short)0, s1, false, false);
      }
    }
    // scale + causal mask + online softmax
#pragma unroll
    for (int r = 0; r < 8; r++) {
      const int grow = q0 + rowOff + r;
      float a0 = s0[r] * inv_norm;
      float a1 = s1[r] * inv_norm;
      if (kbase + rsel > grow) a0 = NEG_INF_;
      if (kbase + rsel + 16 > grow) a1 = NEG_INF_;
      const float rm = red_max16(fmaxf(a0, a1));
      const float mnew = fmaxf(mrow[r], rm);
      const float alpha = __expf(mrow[r] - mnew);
      const float p0 = __expf(a0 - mnew);
      const float p1 = __expf(a1 - mnew);
      lrow[r] = lrow[r] * alpha + red_sum16(p0 + p1);
      mrow[r] = mnew;
#pragma unroll
      for (int f = 0; f < 8; f++) o[f][r] *= alpha;
      Pls[wave][rowOff + r][rsel] = (bf16_t)p0;
      Pls[wave][rowOff + r][rsel + 16] = (bf16_t)p1;
    }
    // P(16x32) x V(32x128)
    {
      Frag pf;
      const int klo = (lane & 16) ? 16 : 0;
      const bf16_t* pp = &Pls[wave][rsel][0];
      pf.u[0] = *(const uint4*)(pp + klo);
      pf.u[1] = *(const uint4*)(pp + klo + 8);
      const int kvlo = (lane & 16) ? 16 : 0;
#pragma unroll
      for (int f = 0; f < 8; f++) {
        Frag bv;
        const bf16_t* vp = &Vt[f * 16 + rsel][0] + kvlo;
        bv.u[0] = *(const uint4*)(vp);
        bv.u[1] = *(const uint4*)(vp + 8);
        o[f] = __builtin_amdgcn_wmma_f32_16x16x32_bf16(false, pf.v, false, bv.v, (short)0, o[f], false, false);
      }
    }
    __syncthreads();
  }

  // epilogue: normalize, gate, store
#pragma unroll
  for (int r = 0; r < 8; r++) {
    const int grow = q0 + rowOff + r;
    const float rl = 1.0f / lrow[r];
    const size_t tokh = ((size_t)(b * S_ + grow) * H_) + h;
#pragma unroll
    for (int f = 0; f < 8; f++) {
      const int d = f * 16 + rsel;
      const float g = (float)qg[tokh * 256 + 128 + d];  // pre-sigmoided gate
      const float val = o[f][r] * rl * g;
      out[(size_t)(b * S_ + grow) * (H_ * HD_) + h * HD_ + d] = (bf16_t)val;
    }
  }
}

// -------------------------------------------------------------------- launcher
extern "C" void kernel_launch(void* const* d_in, const int* in_sizes, int n_in,
                              void* d_out, int out_size, void* d_ws, size_t ws_size,
                              hipStream_t stream) {
  (void)in_sizes; (void)n_in; (void)out_size; (void)ws_size;
  const float* hs = (const float*)d_in[0];
  const float* Wq = (const float*)d_in[1];
  const float* Wk = (const float*)d_in[2];
  const float* Wv = (const float*)d_in[3];
  const float* Wo = (const float*)d_in[4];
  const float* qsc = (const float*)d_in[5];
  const float* ksc = (const float*)d_in[6];
  // d_in[7] = attention_mask (causal; handled analytically)

  char* ws = (char*)d_ws;
  size_t off = 0;
  auto alloc = [&](size_t elems) -> bf16_t* {
    bf16_t* p = (bf16_t*)(ws + off);
    off += ((elems * sizeof(bf16_t)) + 255) & ~(size_t)255;
    return p;
  };
  const size_t BS = (size_t)B_ * S_;  // 4096
  bf16_t* hsB = alloc(BS * M_);                   // 16 MB
  bf16_t* WqB = alloc((size_t)M_ * H_ * 2 * HD_); // 16 MB
  bf16_t* WkB = alloc((size_t)M_ * KV_ * HD_);    //  2 MB
  bf16_t* WvB = alloc((size_t)M_ * KV_ * HD_);    //  2 MB
  bf16_t* WoB = alloc((size_t)H_ * HD_ * M_);     //  8 MB
  bf16_t* qgB = alloc(BS * H_ * 2 * HD_);         // 32 MB (q | sigmoid(gate))
  bf16_t* kB  = alloc(BS * KV_ * HD_);            //  4 MB
  bf16_t* vB  = alloc(BS * KV_ * HD_);            //  4 MB
  bf16_t* atB = alloc(BS * H_ * HD_);             // 16 MB

  cvt_f32_bf16<<<4096, 256, 0, stream>>>(hs, hsB, (int)(BS * M_));
  cvt_f32_bf16<<<4096, 256, 0, stream>>>(Wq, WqB, M_ * H_ * 2 * HD_);
  cvt_f32_bf16<<<1024, 256, 0, stream>>>(Wk, WkB, M_ * KV_ * HD_);
  cvt_f32_bf16<<<1024, 256, 0, stream>>>(Wv, WvB, M_ * KV_ * HD_);
  cvt_f32_bf16<<<2048, 256, 0, stream>>>(Wo, WoB, H_ * HD_ * M_);

  // QKV projections (bf16 WMMA, bf16 out); block tile 128x256
  gemm_bf16_wmma<true><<<dim3(16, 32), 256, 0, stream>>>(hsB, WqB, qgB, 4096, 4096, 2048);
  gemm_bf16_wmma<true><<<dim3(2, 32), 256, 0, stream>>>(hsB, WkB, kB, 4096, 512, 2048);
  gemm_bf16_wmma<true><<<dim3(2, 32), 256, 0, stream>>>(hsB, WvB, vB, 4096, 512, 2048);

  // RMS-norm + partial RoPE (+ gate sigmoid), in place
  epi_norm_rope<true><<<(int)(BS * H_ / 8), 256, 0, stream>>>(qgB, qsc, (int)(BS * H_), 256, H_);
  epi_norm_rope<false><<<(int)(BS * KV_ / 8), 256, 0, stream>>>(kB, ksc, (int)(BS * KV_), 128, KV_);

  // flash attention
  attn_fwd<<<dim3(S_ / 64, H_, B_), 128, 0, stream>>>(qgB, kB, vB, atB);

  // output projection -> fp32 d_out
  gemm_bf16_wmma<false><<<dim3(8, 32), 256, 0, stream>>>(atB, WoB, d_out, 4096, 2048, 2048);
}
